// GreedyViG_6631429505555
// MI455X (gfx1250) — matrix operand order repaired and checked by reading, not compiled, not run on hardware
//
#include <hip/hip_runtime.h>
#include <hip/hip_bf16.h>

// ---------------------------------------------------------------------------
// GreedyViG forward for MI455X (gfx1250, wave32).
// - 1x1 convs: LDS-staged bf16 GEMM, K-step 64 => 4x v_wmma_f32_16x16x32_bf16
//   per barrier pair (32M x 64N tile per 128-thread block), b128 global loads,
//   b128 LDS fragment reads (144B row pitch: aligned + bank-conflict-free).
// - 3x3 convs (stem/downsample/head): implicit-im2col WMMA gather kernel;
//   head c1 fuses the x32 bilinear upsample into its operand load so the
//   201MB upsampled tensor never exists (8x8 source stays L2-resident).
// - BN fold / exact-erf GELU / layer-scale residual fused into epilogues.
// ---------------------------------------------------------------------------

typedef __attribute__((ext_vector_type(16))) __bf16 v16bf;
typedef __attribute__((ext_vector_type(8)))  __bf16 v8bf;
typedef __attribute__((ext_vector_type(4)))  __bf16 v4bf;
typedef __attribute__((ext_vector_type(2)))  __bf16 v2bf;
typedef __attribute__((ext_vector_type(8)))  float  v8f;

#define PW_PITCH 72  // bf16/row: 144B = 9*16B -> aligned b128 reads, 16 distinct banks

__device__ __forceinline__ float gelu_f(float x) {
  return 0.5f * x * (1.0f + erff(x * 0.70710678118654752440f));
}

__device__ __forceinline__ float epilogue_f(float acc, int m,
                                            const float* __restrict__ bias,
                                            const float* __restrict__ bn_g,
                                            const float* __restrict__ bn_b,
                                            const float* __restrict__ bn_m,
                                            const float* __restrict__ bn_v,
                                            int use_gelu) {
  float s = 1.0f, t = 0.0f;
  if (bias) t = bias[m];
  if (bn_g) {
    float sc = bn_g[m] * rsqrtf(bn_v[m] + 1e-5f);
    t = t * sc + bn_b[m] - bn_m[m] * sc;
    s = sc;
  }
  float y = acc * s + t;
  if (use_gelu) y = gelu_f(y);
  return y;
}

// ---------------------------------------------------------------------------
// Pointwise (1x1) conv as LDS-staged WMMA GEMM.
//   out[b,m,n] = sum_k W[m,k] * x[b,k,n],  K=Cin, N=HW (multiple of 64).
// Block = 128 threads (4 waves). Tile: M=32, N=64, K-step=64 (2 WMMA chunks).
// A LDS: [32m][64k] bf16; B LDS: [64n][64k] bf16 (transposed, pair-packed b32
// stores). Each wave: per chunk 1 B fragment x 2 A fragments -> 4 WMMA/step.
// Ragged K/M handled by zero-padding, so EXEC stays uniform around WMMA.
// ---------------------------------------------------------------------------
__global__ void __launch_bounds__(128) pwconv_wmma_kernel(
    const float* __restrict__ x, const float* __restrict__ w,
    const float* __restrict__ bias,
    const float* __restrict__ bn_g, const float* __restrict__ bn_b,
    const float* __restrict__ bn_m, const float* __restrict__ bn_v,
    const float* __restrict__ res, const float* __restrict__ ls,
    float* __restrict__ out,
    int in_cn, int in_coff, int K, int Np,
    int out_cn, int out_coff, int Cout, int use_gelu) {
  __shared__ __bf16 Abuf[32 * PW_PITCH];  // [m][k]
  __shared__ __bf16 Bbuf[64 * PW_PITCH];  // [n][k] (transposed)

  const int b    = blockIdx.x;
  const int tid  = threadIdx.x;
  const int lane = tid & 31;
  const int wave = tid >> 5;
  const int m0   = blockIdx.y * 32;
  const int n0   = blockIdx.z * 64;
  const int g16  = lane >> 4;
  const int lr   = lane & 15;
  const int nn   = n0 + wave * 16 + lr;  // always < Np (Np % 64 == 0)

  const float* xbase = x + ((size_t)b * in_cn + in_coff) * (size_t)Np;

  v8f acc0 = {0.f, 0.f, 0.f, 0.f, 0.f, 0.f, 0.f, 0.f};
  v8f acc1 = {0.f, 0.f, 0.f, 0.f, 0.f, 0.f, 0.f, 0.f};

  for (int kk = 0; kk < K; kk += 64) {
    // ---- stage A: 32 rows(m) x 64 cols(k); 512 float4 slots, 4/thread ----
#pragma unroll
    for (int j = 0; j < 4; ++j) {
      int f   = tid + 128 * j;
      int row = f >> 4;            // m 0..31
      int c4  = (f & 15) << 2;     // k offset 0..60
      int m   = m0 + row;
      int k   = kk + c4;
      float4 wv = make_float4(0.f, 0.f, 0.f, 0.f);
      if (m < Cout && k < K) wv = *(const float4*)&w[(size_t)m * K + k];
      v4bf o;
      o[0] = (__bf16)wv.x; o[1] = (__bf16)wv.y;
      o[2] = (__bf16)wv.z; o[3] = (__bf16)wv.w;
      *(v4bf*)&Abuf[row * PW_PITCH + c4] = o;
    }
    // ---- stage B: 64 rows(k) x 64 cols(n) -> [n][k]; k-pairs packed to b32 ----
#pragma unroll
    for (int j = 0; j < 4; ++j) {
      int f   = tid + 128 * j;     // 512 pair-slots
      int k2  = f >> 4;            // k-pair 0..31
      int n4  = (f & 15) << 2;     // n offset 0..60
      int ka  = kk + k2 * 2;
      int kb2 = ka + 1;
      float4 xa = make_float4(0.f, 0.f, 0.f, 0.f);
      float4 xb = make_float4(0.f, 0.f, 0.f, 0.f);
      if (ka  < K) xa = *(const float4*)&xbase[(size_t)ka  * Np + n0 + n4];
      if (kb2 < K) xb = *(const float4*)&xbase[(size_t)kb2 * Np + n0 + n4];
      v2bf p;
      p[0] = (__bf16)xa.x; p[1] = (__bf16)xb.x;
      *(v2bf*)&Bbuf[(n4 + 0) * PW_PITCH + k2 * 2] = p;
      p[0] = (__bf16)xa.y; p[1] = (__bf16)xb.y;
      *(v2bf*)&Bbuf[(n4 + 1) * PW_PITCH + k2 * 2] = p;
      p[0] = (__bf16)xa.z; p[1] = (__bf16)xb.z;
      *(v2bf*)&Bbuf[(n4 + 2) * PW_PITCH + k2 * 2] = p;
      p[0] = (__bf16)xa.w; p[1] = (__bf16)xb.w;
      *(v2bf*)&Bbuf[(n4 + 3) * PW_PITCH + k2 * 2] = p;
    }
    __syncthreads();

#pragma unroll
    for (int ch = 0; ch < 2; ++ch) {      // two 32-wide WMMA chunks
      const int kb0 = ch * 32;
      // A lane layout (ISA 7.12.2): row=lr, e<8 -> k=g*8+e ; e>=8 -> k=16+g*8+(e-8)
      v8bf a0lo = *(const v8bf*)&Abuf[lr * PW_PITCH + kb0 + g16 * 8];
      v8bf a0hi = *(const v8bf*)&Abuf[lr * PW_PITCH + kb0 + 16 + g16 * 8];
      v8bf a1lo = *(const v8bf*)&Abuf[(lr + 16) * PW_PITCH + kb0 + g16 * 8];
      v8bf a1hi = *(const v8bf*)&Abuf[(lr + 16) * PW_PITCH + kb0 + 16 + g16 * 8];
      // B lane layout: col=lr (of this wave's slice), e<8 -> k=g*8+e ; e>=8 -> k=16+g*8+(e-8)
      int brow = (wave * 16 + lr) * PW_PITCH + kb0 + g16 * 8;
      v8bf blo = *(const v8bf*)&Bbuf[brow];
      v8bf bhi = *(const v8bf*)&Bbuf[brow + 16];

      v16bf a0 = __builtin_shufflevector(a0lo, a0hi, 0, 1, 2, 3, 4, 5, 6, 7,
                                         8, 9, 10, 11, 12, 13, 14, 15);
      v16bf a1 = __builtin_shufflevector(a1lo, a1hi, 0, 1, 2, 3, 4, 5, 6, 7,
                                         8, 9, 10, 11, 12, 13, 14, 15);
      v16bf bm = __builtin_shufflevector(blo, bhi, 0, 1, 2, 3, 4, 5, 6, 7,
                                         8, 9, 10, 11, 12, 13, 14, 15);

      acc0 = __builtin_amdgcn_wmma_f32_16x16x32_bf16(false, a0, false, bm,
                                                     (short)0, acc0, false, false);
      acc1 = __builtin_amdgcn_wmma_f32_16x16x32_bf16(false, a1, false, bm,
                                                     (short)0, acc1, false, false);
    }
    __syncthreads();
  }

#pragma unroll
  for (int r = 0; r < 8; ++r) {
    int mA = m0 + r + 8 * g16;       // acc0 tile
    if (mA < Cout) {
      float y = epilogue_f(acc0[r], mA, bias, bn_g, bn_b, bn_m, bn_v, use_gelu);
      if (res) {
        float l = ls ? ls[mA] : 1.0f;
        y = res[((size_t)b * Cout + mA) * (size_t)Np + nn] + l * y;
      }
      out[((size_t)b * out_cn + out_coff + mA) * (size_t)Np + nn] = y;
    }
    int mB = mA + 16;                // acc1 tile
    if (mB < Cout) {
      float y = epilogue_f(acc1[r], mB, bias, bn_g, bn_b, bn_m, bn_v, use_gelu);
      if (res) {
        float l = ls ? ls[mB] : 1.0f;
        y = res[((size_t)b * Cout + mB) * (size_t)Np + nn] + l * y;
      }
      out[((size_t)b * out_cn + out_coff + mB) * (size_t)Np + nn] = y;
    }
  }
}

// ---------------------------------------------------------------------------
// Generic im2col conv (3x3 paths). Optional fused bilinear-upsampled input.
// ---------------------------------------------------------------------------
__device__ __forceinline__ float sample_in(const float* __restrict__ x, int b,
                                           int in_cn, int ci,
                                           int hi, int wi, int Hin, int Win,
                                           int srcH, int srcW) {
  if (hi < 0 || hi >= Hin || wi < 0 || wi >= Win) return 0.0f;  // conv zero-pad
  if (srcH == 0) {
    return x[(((size_t)b * in_cn + ci) * (size_t)Hin + hi) * (size_t)Win + wi];
  }
  float sy = (hi + 0.5f) * ((float)srcH / (float)Hin) - 0.5f;
  float sx = (wi + 0.5f) * ((float)srcW / (float)Win) - 0.5f;
  sy = fminf(fmaxf(sy, 0.0f), (float)(srcH - 1));
  sx = fminf(fmaxf(sx, 0.0f), (float)(srcW - 1));
  int y0 = (int)floorf(sy);
  int x0 = (int)floorf(sx);
  int y1 = min(y0 + 1, srcH - 1);
  int x1 = min(x0 + 1, srcW - 1);
  float fy = sy - (float)y0, fx = sx - (float)x0;
  const float* base = x + ((size_t)b * in_cn + ci) * (size_t)srcH * srcW;
  float v00 = base[y0 * srcW + x0];
  float v01 = base[y0 * srcW + x1];
  float v10 = base[y1 * srcW + x0];
  float v11 = base[y1 * srcW + x1];
  return v00 * (1.0f - fy) * (1.0f - fx) + v01 * (1.0f - fy) * fx +
         v10 * fy * (1.0f - fx) + v11 * fy * fx;
}

__global__ void conv_wmma_kernel(
    const float* __restrict__ x, const float* __restrict__ w,
    const float* __restrict__ bias,
    const float* __restrict__ bn_g, const float* __restrict__ bn_b,
    const float* __restrict__ bn_m, const float* __restrict__ bn_v,
    float* __restrict__ out,
    int in_cn, int Cin, int Hin, int Win, int srcH, int srcW,
    int Cout, int Hout, int lw /*log2(Wout)*/,
    int ks, int stride, int pad, int use_gelu) {
  const int b    = blockIdx.x;
  const int lane = threadIdx.x & 31;
  const int wave = threadIdx.x >> 5;
  const int m0   = blockIdx.y * 16;
  const int Wout = 1 << lw;
  const int Np   = Hout << lw;
  const int n0   = blockIdx.z * 64 + wave * 16;
  const int g16  = lane >> 4;
  const int mr   = lane & 15;
  const int nn   = n0 + (lane & 15);
  const int K    = Cin * ks * ks;
  const int ho   = nn >> lw;          // Wout is a power of two
  const int wo   = nn & (Wout - 1);
  const bool mrow_ok = (m0 + mr) < Cout;
  const float* wrow = w + (size_t)(m0 + mr) * K;

  v8f acc = {0.f, 0.f, 0.f, 0.f, 0.f, 0.f, 0.f, 0.f};

  for (int kk = 0; kk < K; kk += 32) {
    if (kk + 32 < K && mrow_ok) __builtin_prefetch(&wrow[kk + 32], 0, 0);
    v16bf amat, bmat;
#pragma unroll
    for (int e = 0; e < 16; ++e) {
      int ka = kk + ((e < 8) ? (g16 * 8 + e) : (16 + g16 * 8 + (e - 8)));
      float av = (mrow_ok && ka < K) ? wrow[ka] : 0.0f;
      amat[e] = (__bf16)av;
    }
#pragma unroll
    for (int e = 0; e < 16; ++e) {
      int kb = kk + ((e < 8) ? (g16 * 8 + e) : (16 + g16 * 8 + (e - 8)));
      float bv = 0.0f;
      if (kb < K && nn < Np) {
        int ci, dy, dx;
        if (ks == 3) {               // constant divisors -> mul/shift magic
          ci = kb / 9; int r2 = kb - ci * 9; dy = r2 / 3; dx = r2 - dy * 3;
        } else {
          int kk2 = ks * ks; ci = kb / kk2; int r2 = kb - ci * kk2;
          dy = r2 / ks; dx = r2 - dy * ks;
        }
        int hi = ho * stride + dy - pad;
        int wi = wo * stride + dx - pad;
        bv = sample_in(x, b, in_cn, ci, hi, wi, Hin, Win, srcH, srcW);
      }
      bmat[e] = (__bf16)bv;
    }
    acc = __builtin_amdgcn_wmma_f32_16x16x32_bf16(false, amat, false, bmat,
                                                  (short)0, acc, false, false);
  }

  if (nn >= Np) return;
#pragma unroll
  for (int r = 0; r < 8; ++r) {
    int m = m0 + r + 8 * g16;
    if (m >= Cout) continue;
    float y = epilogue_f(acc[r], m, bias, bn_g, bn_b, bn_m, bn_v, use_gelu);
    out[((size_t)b * Cout + m) * (size_t)Np + nn] = y;
  }
}

// ---------------------------------------------------------------------------
// Depthwise conv (3x3 IR dw with BN+GELU, 7x7 CPE with bias+residual)
// ---------------------------------------------------------------------------
__global__ void dwconv_kernel(const float* __restrict__ x, const float* __restrict__ w,
                              const float* __restrict__ bias,
                              const float* __restrict__ bn_g, const float* __restrict__ bn_b,
                              const float* __restrict__ bn_m, const float* __restrict__ bn_v,
                              const float* __restrict__ res, const float* __restrict__ ls,
                              float* __restrict__ out,
                              int B, int C, int H, int W, int ks, int pad, int use_gelu) {
  size_t tid = (size_t)blockIdx.x * blockDim.x + threadIdx.x;
  size_t total = (size_t)B * C * H * W;
  if (tid >= total) return;
  int wo = (int)(tid % W);
  int ho = (int)((tid / W) % H);
  int c  = (int)((tid / ((size_t)W * H)) % C);
  int b  = (int)(tid / ((size_t)W * H * C));
  const float* xp = x + ((size_t)b * C + c) * (size_t)H * W;
  const float* wp = w + (size_t)c * ks * ks;
  float acc = 0.0f;
  for (int dy = 0; dy < ks; ++dy) {
    int hi = ho + dy - pad;
    if (hi < 0 || hi >= H) continue;
    for (int dx = 0; dx < ks; ++dx) {
      int wi = wo + dx - pad;
      if (wi < 0 || wi >= W) continue;
      acc += xp[hi * W + wi] * wp[dy * ks + dx];
    }
  }
  float y = epilogue_f(acc, c, bias, bn_g, bn_b, bn_m, bn_v, use_gelu);
  if (res) { float l = ls ? ls[c] : 1.0f; y = res[tid] + l * y; }
  out[tid] = y;
}

// ---------------------------------------------------------------------------
// DynamicMRConv4d pieces
// ---------------------------------------------------------------------------
__global__ void dynmr_norm_kernel(const float* __restrict__ x, int in_cn, int coff,
                                  float* __restrict__ nrm, int B, int C, int H, int W) {
  int HW = H * W;
  int tid = blockIdx.x * blockDim.x + threadIdx.x;
  if (tid >= B * HW) return;
  int b = tid / HW, hw = tid % HW;
  int h = hw / W, wv = hw % W;
  int h2 = h - H / 2; if (h2 < 0) h2 += H;
  int w2 = wv - W / 2; if (w2 < 0) w2 += W;
  const float* base = x + ((size_t)b * in_cn + coff) * (size_t)HW;
  float s = 0.0f;
  for (int c = 0; c < C; ++c) {
    float a = base[(size_t)c * HW + hw];
    float r = base[(size_t)c * HW + h2 * W + w2];
    s += fabsf(a - r);
  }
  nrm[tid] = s;
}

__global__ void dynmr_stats_kernel(const float* __restrict__ nrm,
                                   float* __restrict__ thr, int HW) {
  int b = blockIdx.x;
  __shared__ float ssum[256], ssq[256];
  float s = 0.0f, q = 0.0f;
  for (int i = threadIdx.x; i < HW; i += blockDim.x) {
    float v = nrm[b * HW + i];
    s += v; q += v * v;
  }
  ssum[threadIdx.x] = s; ssq[threadIdx.x] = q;
  __syncthreads();
  for (int st = 128; st > 0; st >>= 1) {
    if ((int)threadIdx.x < st) {
      ssum[threadIdx.x] += ssum[threadIdx.x + st];
      ssq[threadIdx.x]  += ssq[threadIdx.x + st];
    }
    __syncthreads();
  }
  if (threadIdx.x == 0) {
    float mean = ssum[0] / (float)HW;
    float var  = (ssq[0] - (float)HW * mean * mean) / (float)(HW - 1);  // ddof=1
    var = fmaxf(var, 0.0f);
    thr[b] = mean - sqrtf(var);
  }
}

__global__ void dynmr_agg_kernel(const float* __restrict__ cat, float* __restrict__ catw,
                                 const float* __restrict__ thr,
                                 int B, int C, int H, int W, int Kstep) {
  int HW = H * W;
  int tid = blockIdx.x * blockDim.x + threadIdx.x;
  if (tid >= B * HW) return;
  int b = tid / HW, hw = tid % HW;
  int h = hw / W, wv = hw % W;
  const float* xb = cat + (size_t)b * (2 * C) * (size_t)HW;
  float* xj = catw + (size_t)b * (2 * C) * (size_t)HW + (size_t)C * HW;
  for (int c = 0; c < C; ++c) xj[(size_t)c * HW + hw] = 0.0f;
  float tv = thr[b];

  for (int axis = 0; axis < 2; ++axis) {
    int lim = (axis == 0) ? H : W;
    for (int s = 0; s < lim; s += Kstep) {
      int rhw;
      if (axis == 0) { int hh = h - s; if (hh < 0) hh += H; rhw = hh * W + wv; }
      else           { int ww = wv - s; if (ww < 0) ww += W; rhw = h * W + ww; }
      float d = 0.0f;
      for (int c = 0; c < C; ++c)
        d += fabsf(xb[(size_t)c * HW + hw] - xb[(size_t)c * HW + rhw]);
      if (d < tv) {
        for (int c = 0; c < C; ++c) {
          float cand = xb[(size_t)c * HW + rhw] - xb[(size_t)c * HW + hw];
          size_t ix = (size_t)c * HW + hw;
          xj[ix] = fmaxf(xj[ix], cand);
        }
      }
    }
  }
}

// ---------------------------------------------------------------------------
// Host orchestration
// ---------------------------------------------------------------------------
extern "C" void kernel_launch(void* const* d_in, const int* in_sizes, int n_in,
                              void* d_out, int out_size, void* d_ws, size_t ws_size,
                              hipStream_t stream) {
  const int CHa[4] = {48, 96, 192, 384};
  const int KSa[4] = {2, 2, 4, 8};
  const int B = in_sizes[0] / (256 * 256);  // = 2

  // ---- parameter walk (setup_inputs insertion order) ----
  int cursor = 1;
  auto nx = [&]() -> const float* { return (const float*)d_in[cursor++]; };
  struct CBP { const float *w, *b, *g, *bb, *m, *v; };
  auto getCB = [&]() { CBP p; p.w = nx(); p.b = nx(); p.g = nx(); p.bb = nx(); p.m = nx(); p.v = nx(); return p; };

  CBP s1 = getCB();            // stem.c1
  CBP s2 = getCB();            // stem.c2
  struct StageP {
    CBP down, pw1, dw, pw2, fc1, gc, fc2, ffn1, ffn2;
    const float *ls, *cpew, *cpeb, *ls1, *ls2;
  };
  StageP sp[4];
  for (int i = 0; i < 4; ++i) {
    if (i > 0) sp[i].down = getCB();
    sp[i].pw1 = getCB(); sp[i].dw = getCB(); sp[i].pw2 = getCB(); sp[i].ls = nx();
    sp[i].cpew = nx(); sp[i].cpeb = nx();
    sp[i].fc1 = getCB(); sp[i].gc = getCB(); sp[i].fc2 = getCB();
    sp[i].ffn1 = getCB(); sp[i].ffn2 = getCB();
    sp[i].ls1 = nx(); sp[i].ls2 = nx();
  }
  CBP h1 = getCB();                                   // head.c1
  const float* h2w = nx(); const float* h2b = nx();   // head.c2

  // ---- workspace bump allocator ----
  float* wp = (float*)d_ws;
  auto alloc = [&](size_t n) { float* r = wp; wp += n; return r; };
  float* t24 = alloc((size_t)B * 24 * 128 * 128);
  const size_t S = (size_t)B * 48 * 64 * 64;         // max per-stage activation
  float* bA = alloc(S); float* bB = alloc(S); float* bC = alloc(S);
  float* cat = alloc(2 * S);                         // [B, 2c, H, W] concat
  const size_t Q = (size_t)B * 192 * 64 * 64;        // max expanded (4c)
  float* q1 = alloc(Q); float* q2 = alloc(Q);
  float* nrm = alloc((size_t)B * 64 * 64);
  float* thr = alloc((size_t)B);
  float* hc1 = alloc((size_t)B * 192 * 256 * 256);   // head c1 activation

  // ---- launch helpers ----
  auto log2i = [](int v) { int l = 0; while ((1 << l) < v) ++l; return l; };

  auto conv3 = [&](const float* x, int in_cn, int Cin, int Hin, int Win,
                   int srcH, int srcW, const CBP* p, int use_gelu,
                   float* out, int Cout, int stride) {
    int Hout = (Hin + 2 - 3) / stride + 1;
    int Wout = (Win + 2 - 3) / stride + 1;
    int Np = Hout * Wout;
    dim3 grid(B, (Cout + 15) / 16, (Np + 63) / 64);
    conv_wmma_kernel<<<grid, 128, 0, stream>>>(
        x, p->w, p->b, p->g, p->bb, p->m, p->v, out,
        in_cn, Cin, Hin, Win, srcH, srcW, Cout, Hout, log2i(Wout), 3, stride, 1,
        use_gelu);
  };
  auto pconv = [&](const float* x, int in_cn, int in_coff, int Cin, int Np,
                   const CBP* p, const float* cw, const float* cb, int use_gelu,
                   const float* res, const float* ls,
                   float* out, int out_cn, int out_coff, int Cout) {
    dim3 grid(B, (Cout + 31) / 32, Np / 64);
    pwconv_wmma_kernel<<<grid, 128, 0, stream>>>(
        x, p ? p->w : cw, p ? p->b : cb,
        p ? p->g : nullptr, p ? p->bb : nullptr, p ? p->m : nullptr, p ? p->v : nullptr,
        res, ls, out, in_cn, in_coff, Cin, Np, out_cn, out_coff, Cout, use_gelu);
  };
  auto dwc = [&](const float* x, const float* w, const float* bias, const CBP* bn,
                 const float* res, const float* ls, int use_gelu,
                 float* out, int C, int H, int ks, int pad) {
    size_t total = (size_t)B * C * H * H;
    dwconv_kernel<<<(unsigned)((total + 255) / 256), 256, 0, stream>>>(
        x, w, bias,
        bn ? bn->g : nullptr, bn ? bn->bb : nullptr, bn ? bn->m : nullptr, bn ? bn->v : nullptr,
        res, ls, out, B, C, H, H, ks, pad, use_gelu);
  };

  // ---- stem ----
  const float* xin = (const float*)d_in[0];
  conv3(xin, 1, 1, 256, 256, 0, 0, &s1, 1, t24, 24, 2);
  conv3(t24, 24, 24, 128, 128, 0, 0, &s2, 1, bA, 48, 2);

  float* cur = bA; float* f1 = bB; float* f2 = bC;
  int Hc = 64, Cp = 48;
  for (int i = 0; i < 4; ++i) {
    int c = CHa[i];
    StageP& st = sp[i];
    if (i > 0) {
      conv3(cur, Cp, Cp, Hc, Hc, 0, 0, &st.down, 0, f1, c, 2);
      Hc /= 2;
      float* t = cur; cur = f1; f1 = t;
    }
    int HW = Hc * Hc;
    // inverted residual: x + ls * pw2(dw(pw1(x)))
    pconv(cur, c, 0, c, HW, &st.pw1, nullptr, nullptr, 1, nullptr, nullptr,
          q1, 4 * c, 0, 4 * c);
    dwc(q1, st.dw.w, st.dw.b, &st.dw, nullptr, nullptr, 1, q2, 4 * c, Hc, 3, 1);
    pconv(q2, 4 * c, 0, 4 * c, HW, &st.pw2, nullptr, nullptr, 0, cur, st.ls,
          f1, c, 0, c);
    { float* t = cur; cur = f1; f1 = t; }
    // graph block
    dwc(cur, st.cpew, st.cpeb, nullptr, cur, nullptr, 0, f1, c, Hc, 7, 3);  // CPE + x
    pconv(f1, c, 0, c, HW, &st.fc1, nullptr, nullptr, 0, nullptr, nullptr,
          cat, 2 * c, 0, c);
    dynmr_norm_kernel<<<(B * HW + 255) / 256, 256, 0, stream>>>(cat, 2 * c, 0, nrm, B, c, Hc, Hc);
    dynmr_stats_kernel<<<B, 256, 0, stream>>>(nrm, thr, HW);
    dynmr_agg_kernel<<<(B * HW + 255) / 256, 256, 0, stream>>>(cat, cat, thr, B, c, Hc, Hc, KSa[i]);
    pconv(cat, 2 * c, 0, 2 * c, HW, &st.gc, nullptr, nullptr, 1, nullptr, nullptr,
          f1, c, 0, c);
    pconv(f1, c, 0, c, HW, &st.fc2, nullptr, nullptr, 0, cur, st.ls1,
          f2, c, 0, c);
    { float* t = cur; cur = f2; f2 = t; }
    pconv(cur, c, 0, c, HW, &st.ffn1, nullptr, nullptr, 1, nullptr, nullptr,
          q1, 4 * c, 0, 4 * c);
    pconv(q1, 4 * c, 0, 4 * c, HW, &st.ffn2, nullptr, nullptr, 0, cur, st.ls2,
          f1, c, 0, c);
    { float* t = cur; cur = f1; f1 = t; }
    Cp = c;
  }

  // ---- head: fused bilinear x32 upsample + 3x3 conv, then 1x1 classifier ----
  conv3(cur, 384, 384, 256, 256, /*srcH*/8, /*srcW*/8, &h1, 1, hc1, 192, 1);
  pconv(hc1, 192, 0, 192, 256 * 256, nullptr, h2w, h2b, 0, nullptr, nullptr,
        (float*)d_out, 150, 0, 150);

  (void)n_in; (void)out_size; (void)ws_size;
}
